// RotateQuantizedLinear_53919019434653
// MI455X (gfx1250) — matrix operand order repaired and checked by reading, not compiled, not run on hardware
//
#include <hip/hip_runtime.h>
#include <hip/hip_bf16.h>

#define IN_F    4096
#define OUT_F   14336
#define GROUPS  128
#define KROT    8
#define BATCH   16
#define QW_ROW  (OUT_F / 8)          // 1792 packed dwords per k-row
#define KSPLIT  16                   // K-dimension split across grid.y
#define KCHUNK  (IN_F / KSPLIT)      // 256 rows per WG (2 quant groups)
#define STAGE_K 64                   // k-rows per LDS stage
#define NSTAGE  (KCHUNK / STAGE_K)   // 4 stages per WG
#define WG_COLS 256                  // 32 dwords = one full 128B line per k-row
#define TILES   4                    // 16-col WMMA tiles per wave

typedef __attribute__((ext_vector_type(16))) _Float16 v16h;
typedef __attribute__((ext_vector_type(2)))  _Float16 h2;
typedef __attribute__((ext_vector_type(8)))  float    v8f;
typedef int v4i __attribute__((vector_size(16)));
typedef __attribute__((address_space(1))) v4i* as1_v4i_ptr;
typedef __attribute__((address_space(3))) v4i* as3_v4i_ptr;

#if __has_builtin(__builtin_amdgcn_global_load_async_to_lds_b128) && \
    __has_builtin(__builtin_amdgcn_s_wait_asynccnt)
#define USE_ASYNC_LDS 1
#endif

// ---------------------------------------------------------------------------
// Kernel 1: Givens-rotation rounds + channel scaling on one batch row in LDS,
// then emit the row pre-swizzled into a contiguous-k WMMA 16-bit A-fragment layout
// (lanes 0-15: K j -> K0..15 ; lanes 16-31: K16..31), so the GEMM
// reads one contiguous v16h per lane per K-step.
// ---------------------------------------------------------------------------
__global__ __launch_bounds__(256)
void rotate_pack_kernel(const _Float16* __restrict__ x,
                        const _Float16* __restrict__ theta,
                        const int*      __restrict__ pairs,
                        const _Float16* __restrict__ cscale,
                        _Float16*       __restrict__ a_frag)
{
    __shared__ _Float16 sx[IN_F];
    const int m   = blockIdx.x;
    const int tid = threadIdx.x;

    for (int i = tid; i < IN_F; i += 256) sx[i] = x[m * IN_F + i];
    __syncthreads();

    for (int k = 0; k < KROT; ++k) {
        for (int p = tid; p < IN_F / 2; p += 256) {   // disjoint pairs per round
            int ia = pairs[k * IN_F + 2 * p];
            int ib = pairs[k * IN_F + 2 * p + 1];
            float t = (float)theta[k * (IN_F / 2) + p];
            _Float16 co = (_Float16)__builtin_cosf(t);
            _Float16 si = (_Float16)__builtin_sinf(t);
            _Float16 a = sx[ia], b = sx[ib];
            sx[ia] = co * a - si * b;
            sx[ib] = si * a + co * b;
        }
        __syncthreads();
    }

    for (int i = tid; i < IN_F; i += 256) sx[i] *= cscale[i];
    __syncthreads();

    for (int idx = tid; idx < IN_F; idx += 256) {
        int s = idx >> 5, e = idx & 31;
        int h = e >> 4, j = e & 15;
        int K = s * 32 + h * 16 + j;
        a_frag[((s * 32 + m + h * 16) << 4) + j] = sx[K];
    }
}

// ---------------------------------------------------------------------------
// Kernel 2: fused int4-dequant GEMM partial. Grid = (56 col-blocks, 16 k-chunks).
// WG: 4 waves x 64 cols each (4 WMMA tiles). qweight streamed in 64-row x 256-col
// stages (8KB = one full 128B line per k-row) via async global->LDS DMA,
// triple-buffered, 2 stages in flight, one barrier per stage.
// ---------------------------------------------------------------------------
__global__ __launch_bounds__(128)
void w4_gemm_wmma(const _Float16* __restrict__ a_frag,
                  const int*      __restrict__ qweight,
                  const int*      __restrict__ qzeros,
                  const _Float16* __restrict__ scales,
                  float*          __restrict__ part)
{
    __shared__ __align__(16) unsigned lds_w[3 * STAGE_K * 32];  // 3 x 8KB stages

    const int tid    = threadIdx.x;
    const int lane   = tid & 31;
    const int wave   = tid >> 5;                 // 0..3
    const int bx     = blockIdx.x;               // 0..55 column block
    const int cz     = blockIdx.y;               // 0..15 k chunk
    const int kb0    = cz * KCHUNK;
    const int colT   = lane & 15;                // column within a 16-tile
    const int hiHalf = (lane >> 4) & 1;          // B: lanes>=16 take K 16..31
    const int sh     = 4 * (lane & 7);           // nibble shift for this column

    // per-stage fetch: 2 threads per k-row, 64B each (half a 128B row slice)
    const int frow  = tid >> 1;                  // 0..63
    const int fhalf = tid & 1;                   // 0..1
    const size_t gbase = (size_t)(kb0 + frow) * QW_ROW + bx * 32 + fhalf * 16;
    const int    lbase = frow * 32 + fhalf * 16; // dword index within a stage

    v8f acc[TILES];
    h2  zb[TILES], sc2[TILES];
    #pragma unroll
    for (int t = 0; t < TILES; ++t) { acc[t] = {}; zb[t] = {}; sc2[t] = {}; }

#ifdef USE_ASYNC_LDS
    #define ISSUE_STAGE(s)                                                          \
    do {                                                                            \
        const int* gp = qweight + gbase + (size_t)(s) * STAGE_K * QW_ROW;           \
        unsigned*  lp = &lds_w[((s) % 3) * (STAGE_K * 32) + lbase];                 \
        __builtin_amdgcn_global_load_async_to_lds_b128(                             \
            (as1_v4i_ptr)(gp + 0),  (as3_v4i_ptr)(lp + 0),  0, 0);                  \
        __builtin_amdgcn_global_load_async_to_lds_b128(                             \
            (as1_v4i_ptr)(gp + 4),  (as3_v4i_ptr)(lp + 4),  0, 0);                  \
        __builtin_amdgcn_global_load_async_to_lds_b128(                             \
            (as1_v4i_ptr)(gp + 8),  (as3_v4i_ptr)(lp + 8),  0, 0);                  \
        __builtin_amdgcn_global_load_async_to_lds_b128(                             \
            (as1_v4i_ptr)(gp + 12), (as3_v4i_ptr)(lp + 12), 0, 0);                  \
    } while (0)
#endif

    auto compute_stage = [&](int s) {
        const unsigned* buf = &lds_w[(s % 3) * (STAGE_K * 32)];
        #pragma unroll
        for (int h = 0; h < 2; ++h) {
            const int kl  = s * STAGE_K + h * 32;    // k within chunk
            const int kkg = kb0 + kl;                // global k
            if ((kkg & (GROUPS - 1)) == 0) {         // wave-uniform group reload
                int g = kkg >> 7;
                #pragma unroll
                for (int t = 0; t < TILES; ++t) {
                    int col = bx * WG_COLS + wave * 64 + t * 16 + colT;
                    _Float16 sc = scales[(size_t)g * OUT_F + col];
                    unsigned zw = (unsigned)qzeros[g * QW_ROW + (col >> 3)];
                    int iz = (zw >> (4 * (col & 7))) & 15;
                    _Float16 z = (_Float16)(-1024.0f - (float)iz);
                    zb[t].x = z;   zb[t].y = z;
                    sc2[t].x = sc; sc2[t].y = sc;
                }
            }
            // one A fragment per lane, shared by all 4 tiles (L2-resident)
            v16h a = *(const v16h*)(a_frag + ((((kkg >> 5) * 32) + lane) << 4));
            const int rl = (h * 32 + hiHalf * 16) * 32;  // stage-local row base (dwords)
            #pragma unroll
            for (int t = 0; t < TILES; ++t) {
                const int wd = wave * 8 + t * 2 + (colT >> 3);
                v16h b = {};
                #pragma unroll
                for (int j = 0; j < 8; ++j) {
                    unsigned w0 = buf[rl + (2 * j) * 32 + wd];
                    unsigned w1 = buf[rl + (2 * j + 1) * 32 + wd];
                    unsigned e0 = (w0 >> sh) & 15u;
                    unsigned e1 = (w1 >> sh) & 15u;
                    unsigned packed = e0 | (e1 << 16) | 0x64006400u; // fp16{1024+e}
                    h2 hp = __builtin_bit_cast(h2, packed);
                    h2 v  = (hp + zb[t]) * sc2[t];
                    b[2 * j]     = v.x;
                    b[2 * j + 1] = v.y;
                }
                acc[t] = __builtin_amdgcn_wmma_f32_16x16x32_f16(
                             false, a, false, b, (short)0, acc[t], false, false);
            }
        }
    };

#ifdef USE_ASYNC_LDS
    ISSUE_STAGE(0);
    ISSUE_STAGE(1);
    for (int s = 0; s < NSTAGE; ++s) {
        if (s < NSTAGE - 1) __builtin_amdgcn_s_wait_asynccnt(4);  // stage s landed
        else                __builtin_amdgcn_s_wait_asynccnt(0);
        __syncthreads();
        compute_stage(s);
        if (s + 2 < NSTAGE) ISSUE_STAGE(s + 2);  // barrier above fenced last readers
    }
#else
    for (int s = 0; s < NSTAGE; ++s) {
        const int* gp = qweight + gbase + (size_t)s * STAGE_K * QW_ROW;
        uint4 g0 = *(const uint4*)(gp + 0);
        uint4 g1 = *(const uint4*)(gp + 4);
        uint4 g2 = *(const uint4*)(gp + 8);
        uint4 g3 = *(const uint4*)(gp + 12);
        __syncthreads();
        uint4* lp = (uint4*)&lds_w[(s % 3) * (STAGE_K * 32) + lbase];
        lp[0] = g0; lp[1] = g1; lp[2] = g2; lp[3] = g3;
        __syncthreads();
        compute_stage(s);
    }
#endif

    // D layout: VGPR r -> M = r (+8 for lanes>=16); N = lane&15
    #pragma unroll
    for (int t = 0; t < TILES; ++t) {
        int col = bx * WG_COLS + wave * 64 + t * 16 + colT;
        #pragma unroll
        for (int r = 0; r < 8; ++r) {
            int mOut = r + hiHalf * 8;
            part[((size_t)cz * BATCH + mOut) * OUT_F + col] = acc[t][r];
        }
    }
}

// ---------------------------------------------------------------------------
// Kernel 3: fold the KSPLIT float partials into fp16 output.
// ---------------------------------------------------------------------------
__global__ __launch_bounds__(256)
void reduce_out_kernel(const float* __restrict__ part, _Float16* __restrict__ out)
{
    int i = blockIdx.x * 256 + threadIdx.x;
    if (i < BATCH * OUT_F) {
        float s = 0.0f;
        #pragma unroll
        for (int c = 0; c < KSPLIT; ++c) s += part[(size_t)c * (BATCH * OUT_F) + i];
        out[i] = (_Float16)s;
    }
}

extern "C" void kernel_launch(void* const* d_in, const int* in_sizes, int n_in,
                              void* d_out, int out_size, void* d_ws, size_t ws_size,
                              hipStream_t stream)
{
    const _Float16* x       = (const _Float16*)d_in[0];
    const _Float16* theta   = (const _Float16*)d_in[1];
    const int*      pairs   = (const int*)d_in[2];
    const _Float16* cscale  = (const _Float16*)d_in[3];
    const int*      qweight = (const int*)d_in[4];
    const int*      qzeros  = (const int*)d_in[5];
    const _Float16* scales  = (const _Float16*)d_in[6];

    _Float16* a_frag = (_Float16*)d_ws;                                   // 128 KB
    float*    part   = (float*)((char*)d_ws + (size_t)BATCH * IN_F * 2);  // 14.7 MB

    rotate_pack_kernel<<<BATCH, 256, 0, stream>>>(x, theta, pairs, cscale, a_frag);
    w4_gemm_wmma<<<dim3(OUT_F / WG_COLS, KSPLIT), 128, 0, stream>>>(
        a_frag, qweight, qzeros, scales, part);
    reduce_out_kernel<<<(BATCH * OUT_F + 255) / 256, 256, 0, stream>>>(
        part, (_Float16*)d_out);
}
